// CharRNN_72791105733304
// MI455X (gfx1250) — matrix operand order repaired and checked by reading, not compiled, not run on hardware
//
#include <hip/hip_runtime.h>
#include <hip/hip_bf16.h>
#include <stdint.h>

// CharRNN GRU (reset_after=True) for MI455X / gfx1250, wave32 + WMMA bf16.
// B=4096, T=256, V=256, H=20, 3H=60, L=15.

typedef __bf16 v16bf __attribute__((ext_vector_type(16)));
typedef float  v8f   __attribute__((ext_vector_type(8)));

#define B_    4096
#define T_    256
#define V_    256
#define H_    20
#define H3_   60
#define L_    15
#define ROWS  16
#define HSTR  33   // padded LDS stride for h (cols 0..31 used, zero-padded K)
#define USTR  65   // padded LDS stride for hU (cols 0..63)

// Fast hardware reciprocal (v_rcp_f32) instead of IEEE divide chains.
__device__ __forceinline__ float fast_rcp(float x) {
#if defined(__AMDGCN__) && __has_builtin(__builtin_amdgcn_rcpf)
    return __builtin_amdgcn_rcpf(x);
#else
    return 1.0f / x;
#endif
}

#if defined(__AMDGCN__) && __has_builtin(__builtin_amdgcn_global_load_async_to_lds_b128)
#define ASYNC_GATHER 1
// Exact parameter types per hipcc diagnostic: int __vector_size(16) pointers,
// global (AS1) source, LDS (AS3) destination.
typedef int v4i_raw __attribute__((vector_size(16)));
typedef __attribute__((address_space(1))) v4i_raw* async_gptr;
typedef __attribute__((address_space(3))) v4i_raw* async_lptr;
#else
#define ASYNC_GATHER 0
#endif

__device__ __forceinline__ void wait_async_lds() {
#if defined(__AMDGCN__)
#if __has_builtin(__builtin_amdgcn_s_wait_asynccnt)
    __builtin_amdgcn_s_wait_asynccnt(0);
#elif ASYNC_GATHER
    asm volatile("s_wait_asynccnt 0x0" ::: "memory");
#endif
#endif
}

__global__ __launch_bounds__(32)
void charrnn_gru_wmma_kernel(const int* __restrict__ xin,
                             const float* __restrict__ dropp,
                             const float* __restrict__ kern,   // [V][3H]
                             const float* __restrict__ rk,     // [H][3H]
                             const float* __restrict__ bias,   // [2][3H]
                             const float* __restrict__ dw,     // [H][L]
                             const float* __restrict__ db,     // [L]
                             float* __restrict__ out)          // [B][L]
{
    __shared__ float sH [ROWS * HSTR];      // h state, f32, zero-padded to K=32
    __shared__ float sHU[ROWS * USTR];      // h @ U + br staging (N=0..63)
    __shared__ float sXW[ROWS * H3_];       // gathered embedding rows for this step
    __shared__ float sBi[H3_];              // input bias bi
    __shared__ float sDw[H_ * L_];
    __shared__ float sDb[L_];

    const int lane = threadIdx.x;           // 0..31 (wave32)
    const int half = lane >> 4;             // lane group 0 / 1
    const int m    = lane & 15;             // row / column within 16-wide tile
    const int row0 = blockIdx.x * ROWS;     // first batch row of this tile

    // ---- one-time setup: zero h (incl. K pad), stage biases / dense weights ----
    for (int i = lane; i < ROWS * HSTR; i += 32) sH[i] = 0.0f;
    for (int i = lane; i < H3_;         i += 32) sBi[i] = bias[i];          // bi
    for (int i = lane; i < H_ * L_;     i += 32) sDw[i] = dw[i];
    if (lane < L_) sDb[lane] = db[lane];

    // ---- pre-pack B tiles of recurrent_kernel (bf16, K padded 20->32, N 60->64),
    //      and C-init tiles broadcasting br[n] (recurrent bias) ----
    // B layout (32x16, 16-bit): lanes 0-15: N=lane, v0-3 K=0..7, v4-7 K=16..23;
    //                           lanes 16-31: N=lane-16, v0-3 K=8..15, v4-7 K=24..31.
    v16bf bU[4];
    v8f   cin[4];
#pragma unroll
    for (int t = 0; t < 4; ++t) {
        const int n = t * 16 + m;
#pragma unroll
        for (int j = 0; j < 8; ++j) {
#pragma unroll
            for (int e = 0; e < 2; ++e) {
                const int k = ((j < 4) ? (half * 8 + 2 * j) : (16 + half * 8 + 2 * (j - 4))) + e;
                float uv = 0.0f;
                if (k < H_ && n < H3_) uv = rk[k * H3_ + n];
                bU[t][2 * j + e] = (__bf16)uv;
            }
        }
        const float bv = (n < H3_) ? bias[H3_ + n] : 0.0f;   // br[n]
#pragma unroll
        for (int r = 0; r < 8; ++r) cin[t][r] = bv;
    }
    __syncthreads();

    // ================= sequential GRU over T steps =================
    for (int ts = 0; ts < T_; ++ts) {
        // 1) per-row embedding index for this step (lanes 0..15 own one row each)
        int xi = 0;
        if (lane < 16) xi = xin[(row0 + lane) * T_ + ts];

        // 2) gather 16 embedding rows (60 f32 = 15 float4 each) into LDS staging.
        //    Table is 60 KB -> permanently L2-resident.  Use the gfx1250 async
        //    global->LDS path (ASYNCcnt) when available so the gather overlaps
        //    the A-pack + WMMA below; otherwise classic VGPR round-trip.
#pragma unroll
        for (int it = 0; it < 8; ++it) {
            const int j = lane + it * 32;            // 0..255, 240 valid
            if (j < 240) {
                const int row = j / 15;
                const int seg = j - row * 15;
                const int idx = __shfl(xi, row);     // broadcast row's index
#if ASYNC_GATHER
                async_gptr gsrc = (async_gptr)(uintptr_t)
                        (((const float4*)kern) + idx * 15 + seg);
                async_lptr ldst = (async_lptr)(uint32_t)(uintptr_t)
                        (((float4*)(sXW + row * H3_)) + seg);
                __builtin_amdgcn_global_load_async_to_lds_b128(gsrc, ldst, 0, 0);
#else
                const float4 v = ((const float4*)kern)[idx * 15 + seg];
                ((float4*)(sXW + row * H3_))[seg] = v;
#endif
            }
        }

        // 3) pack A = h tile (16x32 bf16).  A layout (16-bit, 16x32):
        //    lanes 0-15:  v0-3 K=0..7,  v4-7 K=16..23  (row M=lane)
        //    lanes 16-31: v0-3 K=8..15, v4-7 K=24..31  (row M=lane-16)
        v16bf a;
#pragma unroll
        for (int j = 0; j < 8; ++j) {
            const int kb = (j < 4) ? (half * 8 + 2 * j)
                                   : (16 + half * 8 + 2 * (j - 4));
            a[2 * j]     = (__bf16)sH[m * HSTR + kb];
            a[2 * j + 1] = (__bf16)sH[m * HSTR + kb + 1];
        }

        // 4) hU = h @ U + br : four 16x16x32 bf16 WMMAs, scatter D to LDS.
        //    D layout: VGPR r -> M = r + half*8, N = m + 16*t.
#pragma unroll
        for (int t = 0; t < 4; ++t) {
            v8f d = __builtin_amdgcn_wmma_f32_16x16x32_bf16(
                        false, a, false, bU[t], (short)0, cin[t], false, false);
#pragma unroll
            for (int r = 0; r < 8; ++r)
                sHU[(r + half * 8) * USTR + t * 16 + m] = d[r];
        }
        wait_async_lds();       // embedding tiles resident in LDS
        __syncthreads();

        // 5) gates: 320 (b,n) elements, 10 per lane.
        //    lane -> b = lane&15, n in [half*10, half*10+10)
        {
            const int b  = m;
            const int nb = half * 10;
#pragma unroll
            for (int i = 0; i < 10; ++i) {
                const int n = nb + i;
                const float xz = sXW[b * H3_ + n]          + sBi[n];
                const float xr = sXW[b * H3_ + H_ + n]     + sBi[H_ + n];
                const float xh = sXW[b * H3_ + 2 * H_ + n] + sBi[2 * H_ + n];
                const float hz = sHU[b * USTR + n];
                const float hr = sHU[b * USTR + H_ + n];
                const float hh = sHU[b * USTR + 2 * H_ + n];
                const float z  = fast_rcp(1.0f + __expf(-(xz + hz)));
                const float r  = fast_rcp(1.0f + __expf(-(xr + hr)));
                const float th = xh + r * hh;
                const float hc = 1.0f - 2.0f * fast_rcp(__expf(2.0f * th) + 1.0f);
                const float hp = sH[b * HSTR + n];
                sH[b * HSTR + n] = hc + z * (hp - hc);   // z*hp + (1-z)*hc
            }
        }
        __syncthreads();
    }

    // ---- dropout (rate==0 -> identity; general scale kept) + dense to logits ----
    const float drop  = dropp[0];
    const float scale = (drop > 0.0f) ? fast_rcp(1.0f - drop) : 1.0f;
#pragma unroll
    for (int it = 0; it < 8; ++it) {
        const int e = lane + it * 32;        // 240 outputs per tile
        if (e < 240) {
            const int b = e / L_;
            const int l = e - b * L_;
            float acc = 0.0f;
#pragma unroll
            for (int k = 0; k < H_; ++k)
                acc += sH[b * HSTR + k] * sDw[k * L_ + l];
            out[(row0 + b) * L_ + l] = sDb[l] + scale * acc;
        }
    }
}

extern "C" void kernel_launch(void* const* d_in, const int* in_sizes, int n_in,
                              void* d_out, int out_size, void* d_ws, size_t ws_size,
                              hipStream_t stream) {
    const int*   x    = (const int*)  d_in[0];
    const float* drop = (const float*)d_in[1];
    const float* kern = (const float*)d_in[2];
    const float* rk   = (const float*)d_in[3];
    const float* bias = (const float*)d_in[4];
    const float* dw   = (const float*)d_in[5];
    const float* db   = (const float*)d_in[6];
    float* out = (float*)d_out;
    (void)in_sizes; (void)n_in; (void)out_size; (void)d_ws; (void)ws_size;

    dim3 grid(B_ / ROWS);   // 256 independent 16-row GRU tiles
    dim3 block(32);         // one wave32 per tile
    hipLaunchKernelGGL(charrnn_gru_wmma_kernel, grid, block, 0, stream,
                       x, drop, kern, rk, bias, dw, db, out);
}
